// CRF_87935160418672
// MI455X (gfx1250) — compile-verified, matrix operand order checked by reading, not run on hardware
//
#include <hip/hip_runtime.h>
#include <hip/hip_bf16.h>
#include <math.h>

typedef __attribute__((ext_vector_type(16))) _Float16 v16h;
typedef __attribute__((ext_vector_type(8)))  float    v8f;

#define TT 512
#define CC 256
#define ROWS 16   // batch rows per workgroup (one WMMA M tile)

#define LOG2E 1.4426950408889634f
#define LN2   0.6931471805599453f

// raw v_exp_f32: 2^x
#define EXP2F(x) __builtin_amdgcn_exp2f(x)

// ---------------------------------------------------------------------------
// Forward (log-partition) kernel: exp-space forward recurrence with per-row
// rescaling.  grid = B/16 blocks, 512 threads (16 wave32's).
// Wave w owns batch row w for the max/exp phase and output columns
// [16w,16w+16) for the WMMA phase; its exp(transitions) B-slice stays
// resident in 64 VGPRs (8 x v16h) across the whole T loop.
// ---------------------------------------------------------------------------
__global__ __launch_bounds__(512) void crf_forward_kernel(
    const float* __restrict__ em, const unsigned char* __restrict__ mask,
    const float* __restrict__ trans, const float* __restrict__ startT,
    const float* __restrict__ endT, float* __restrict__ logZ)
{
  __shared__ float    s_alpha[ROWS][CC];   // rescaled log-alpha (16 KB)
  __shared__ _Float16 s_p[ROWS][CC];       // exp(alpha - m) as f16 (8 KB)
  __shared__ float    s_offs[ROWS];        // accumulated rescale offsets

  const int tid  = threadIdx.x;
  const int w    = tid >> 5;               // wave id == row id == N tile id
  const int lane = tid & 31;
  const int half = lane >> 4;              // K-interleave select (WMMA layout)
  const int nl   = lane & 15;
  const int b0   = blockIdx.x * ROWS;

  // ---- B matrix: exp(transitions) slice for this wave, WMMA 16-bit layout.
  v16h breg[8];
  {
    const int n = w * 16 + nl;
#pragma unroll
    for (int kc = 0; kc < 8; ++kc) {
#pragma unroll
      for (int v = 0; v < 8; ++v) {
        const int k0 = kc * 32 + (v >> 2) * 16 + half * 8 + (v & 3) * 2;
        breg[kc][2 * v] =
            (_Float16)EXP2F(trans[(size_t)(k0) * CC + n] * LOG2E);
        breg[kc][2 * v + 1] =
            (_Float16)EXP2F(trans[(size_t)(k0 + 1) * CC + n] * LOG2E);
      }
    }
  }

  // ---- init: alpha_hat = start + em[:,0,:], offsets = 0
  {
    const int idx = tid * 8;               // 512 threads * 8 = 16*256 elements
    const int r = idx >> 8, c = idx & 255;
#pragma unroll
    for (int i = 0; i < 8; ++i)
      s_alpha[r][c + i] =
          startT[c + i] + em[(((size_t)(b0 + r)) * TT + 0) * CC + c + i];
    if (tid < ROWS) s_offs[tid] = 0.0f;
  }
  __syncthreads();

  for (int t = 1; t < TT; ++t) {
    // (A) wave w: row-w max (shfl reduce) + p = exp(alpha - m) -> f16 LDS
    {
      const int j0 = lane * 8;             // 32 lanes * 8 = 256 cols
      float av[8];
      float mx = -3.0e38f;
#pragma unroll
      for (int i = 0; i < 8; ++i) {
        av[i] = s_alpha[w][j0 + i];
        mx = fmaxf(mx, av[i]);
      }
#pragma unroll
      for (int d = 16; d >= 1; d >>= 1) mx = fmaxf(mx, __shfl_xor(mx, d, 32));
      if (lane == 0 && mask[(size_t)(b0 + w) * TT + t]) s_offs[w] += mx;
#pragma unroll
      for (int i = 0; i < 8; ++i)
        s_p[w][j0 + i] = (_Float16)EXP2F((av[i] - mx) * LOG2E);
    }
    __syncthreads();

    // (B) hoist emissions + mask loads so WMMA/LDS work hides HBM latency
    float         emr[8];
    unsigned char mkr[8];
    const int n = w * 16 + nl;
#pragma unroll
    for (int r = 0; r < 8; ++r) {
      const int m = r + 8 * half;          // D layout: VGPR r, lane half -> M
      const size_t eoff = (((size_t)(b0 + m)) * TT + t) * CC + n;
      emr[r] = em[eoff];
      mkr[r] = mask[(size_t)(b0 + m) * TT + t];
      if (r == 0 && t + 1 < TT)
        __builtin_prefetch(&em[eoff + CC], 0, 0);   // next step's emissions
    }

    // acc(16x16) = P(16x256) x expT(256 x 16-slice), K-loop of 8 WMMAs
    v8f acc = {};
#pragma unroll
    for (int kc = 0; kc < 8; ++kc) {
      v16h a;
      const int m = nl;                    // lanes 16-31 mirror M=0..15
#pragma unroll
      for (int v = 0; v < 8; ++v) {
        const int k0 = kc * 32 + (v >> 2) * 16 + half * 8 + (v & 3) * 2;
        a[2 * v]     = s_p[m][k0];
        a[2 * v + 1] = s_p[m][k0 + 1];
      }
      acc = __builtin_amdgcn_wmma_f32_16x16x32_f16(
          false, a, false, breg[kc], (short)0, acc, false, false);
    }

    // alpha' = em + log(acc), masked write-back (acc >= ~0.9, no denorm path)
#pragma unroll
    for (int r = 0; r < 8; ++r) {
      const int m = r + 8 * half;
      const float nv = emr[r] + __log2f(acc[r]) * LN2;
      if (mkr[r]) s_alpha[m][n] = nv;
    }
    __syncthreads();
  }

  // ---- logZ[b] = offs + logsumexp_j(alpha + end); wave w handles row w
  {
    float mx = -3.0e38f;
    for (int j = lane; j < CC; j += 32) mx = fmaxf(mx, s_alpha[w][j] + endT[j]);
#pragma unroll
    for (int d = 16; d >= 1; d >>= 1) mx = fmaxf(mx, __shfl_xor(mx, d, 32));
    float sm = 0.0f;
    for (int j = lane; j < CC; j += 32)
      sm += EXP2F((s_alpha[w][j] + endT[j] - mx) * LOG2E);
#pragma unroll
    for (int d = 16; d >= 1; d >>= 1) sm += __shfl_xor(sm, d, 32);
    if (lane == 0) logZ[b0 + w] = s_offs[w] + mx + __log2f(sm) * LN2;
  }
}

// ---------------------------------------------------------------------------
// Gold sequence score: one block per batch row.
// ---------------------------------------------------------------------------
__global__ __launch_bounds__(256) void crf_gold_kernel(
    const float* __restrict__ em, const long long* __restrict__ tags,
    const unsigned char* __restrict__ mask, const float* __restrict__ trans,
    const float* __restrict__ startT, const float* __restrict__ endT,
    float* __restrict__ gold)
{
  __shared__ float s_sum[256];
  __shared__ int   s_cnt[256];
  const int b = blockIdx.x, tid = threadIdx.x;
  float acc = 0.0f; int cnt = 0;
  for (int t = tid; t < TT; t += 256) {
    const int tg = (int)tags[(size_t)b * TT + t];
    const unsigned char mk = mask[(size_t)b * TT + t];
    const float e = em[(((size_t)b) * TT + t) * CC + tg];
    cnt += mk ? 1 : 0;
    if (t == 0) {
      acc += startT[tg] + e;
    } else {
      const int tp = (int)tags[(size_t)b * TT + t - 1];
      acc += mk ? (trans[(size_t)tp * CC + tg] + e) : 0.0f;
    }
  }
  s_sum[tid] = acc; s_cnt[tid] = cnt;
  __syncthreads();
  for (int s = 128; s >= 1; s >>= 1) {
    if (tid < s) { s_sum[tid] += s_sum[tid + s]; s_cnt[tid] += s_cnt[tid + s]; }
    __syncthreads();
  }
  if (tid == 0) {
    const int len = s_cnt[0] - 1;                  // lengths = sum(mask) - 1
    const int last = (int)tags[(size_t)b * TT + len];
    gold[b] = s_sum[0] + endT[last];
  }
}

// ---------------------------------------------------------------------------
// Final scalar: mean(logZ - gold)
// ---------------------------------------------------------------------------
__global__ __launch_bounds__(128) void crf_reduce_kernel(
    const float* __restrict__ logZ, const float* __restrict__ gold,
    float* __restrict__ out, int B)
{
  __shared__ float s[128];
  const int tid = threadIdx.x;
  s[tid] = (tid < B) ? (logZ[tid] - gold[tid]) : 0.0f;
  __syncthreads();
  for (int d = 64; d >= 1; d >>= 1) {
    if (tid < d) s[tid] += s[tid + d];
    __syncthreads();
  }
  if (tid == 0) out[0] = s[0] / (float)B;
}

extern "C" void kernel_launch(void* const* d_in, const int* in_sizes, int n_in,
                              void* d_out, int out_size, void* d_ws, size_t ws_size,
                              hipStream_t stream) {
  const float*         em     = (const float*)d_in[0];
  const long long*     tags   = (const long long*)d_in[1];
  const unsigned char* mask   = (const unsigned char*)d_in[2];
  const float*         trans  = (const float*)d_in[3];
  const float*         startT = (const float*)d_in[4];
  const float*         endT   = (const float*)d_in[5];

  const int B = in_sizes[1] / TT;          // 128

  float* gold = (float*)d_ws;
  float* logZ = gold + B;

  crf_gold_kernel<<<B, 256, 0, stream>>>(em, tags, mask, trans, startT, endT, gold);
  crf_forward_kernel<<<B / ROWS, 512, 0, stream>>>(em, mask, trans, startT, endT, logZ);
  crf_reduce_kernel<<<1, 128, 0, stream>>>(logZ, gold, (float*)d_out, B);
}
